// RandomProjectionQuantizer_13520557047944
// MI455X (gfx1250) — matrix operand order repaired and checked by reading, not compiled, not run on hardware
//
#include <hip/hip_runtime.h>
#include <stdint.h>

// ---------------------------------------------------------------------------
// RandomProjectionQuantizer fused kernel for MI455X (gfx1250, wave32, WMMA)
//   B=8 N=2048 D=1024 H=8 C=1024 E=64  -> M = B*N = 16384 tokens
//
// projT = W_h x xn^T ; simT = cbn_h x projT (fused argmax).
// Standard (table) K order on the e (contraction) axis of GEMM2 makes the GEMM1
// accumulators directly reinterpretable as GEMM2 B-fragments (no shuffles)
// with codebook A-fragments loaded as two 16-byte chunks (ld_frag_a).
// ---------------------------------------------------------------------------

typedef __attribute__((ext_vector_type(16))) __bf16 v16bf;
typedef __attribute__((ext_vector_type(8)))  float  v8f;

#define DD   1024
#define HH   8
#define EE   64
#define CC   1024
#define TOKW 64          // tokens per workgroup
#define MSUB 4           // 16-token M subtiles per wave
#define XNS  1032        // LDS stride (halves) for xn tile (pad vs bank conflicts)
#define SMEM_MAIN (TOKW * XNS * 2)        // 132096 B

// ---- bf16 helpers (round-to-nearest-even) ---------------------------------
static __device__ __forceinline__ unsigned short f2bf(float f) {
  unsigned u = __builtin_bit_cast(unsigned, f);
  u += 0x7fffu + ((u >> 16) & 1u);
  return (unsigned short)(u >> 16);
}

#if __has_builtin(__builtin_amdgcn_cvt_pk_bf16_f32)
typedef __attribute__((ext_vector_type(2))) __bf16 v2bf;
static __device__ __forceinline__ unsigned pk2bf(float a, float b) {
  return __builtin_bit_cast(unsigned, __builtin_amdgcn_cvt_pk_bf16_f32(a, b));
}
#else
// hardware packed convert: D[15:0]=bf16(S0), D[31:16]=bf16(S1)
static __device__ __forceinline__ unsigned pk2bf(float a, float b) {
  unsigned r;
  asm("v_cvt_pk_bf16_f32 %0, %1, %2" : "=v"(r) : "v"(a), "v"(b));
  return r;
}
#endif

// A-fragment (16x32 bf16), standard K order: lane holds K {0..7,16..23} (lo
// half) / {8..15,24..31}. caller pre-offsets p by (lane>=16 ? 8 : 0).
static __device__ __forceinline__ v16bf ld_frag_a(const __bf16* p) {
  union { uint4 q[2]; v16bf v; } u;
  u.q[0] = *(const uint4*)(p);
  u.q[1] = *(const uint4*)(p + 16);
  return u.v;
}
// Contiguous 32-byte fragment (lanes 0-15 K=0..15,
// lanes 16-31 K=16..31); not called by the kernels in this file.
// caller pre-offsets p by (lane>=16 ? 16 : 0).
static __device__ __forceinline__ v16bf ld_frag_b(const __bf16* p) {
  union { uint4 q[2]; v16bf v; } u;
  u.q[0] = *(const uint4*)(p);
  u.q[1] = *(const uint4*)(p + 8);
  return u.v;
}

static __device__ __forceinline__ v8f wmma_bf16(v16bf a, v16bf b, v8f c) {
  return __builtin_amdgcn_wmma_f32_16x16x32_bf16(false, a, false, b, (short)0, c,
                                                 false, false);
}

// ---------------------------------------------------------------------------
// Prep 1: rand_projs [H][D][E] f32  ->  W_t [H][E][D] bf16 (transposed)
// ---------------------------------------------------------------------------
__global__ void __launch_bounds__(256) prep_w_kernel(const float* __restrict__ rp,
                                                     unsigned short* __restrict__ wt) {
  const int idx = blockIdx.x * 256 + threadIdx.x;      // flat index into [h][e][d]
  const int d = idx & (DD - 1);
  const int e = (idx >> 10) & (EE - 1);
  const int h = idx >> 16;                             // D*E = 65536
  wt[idx] = f2bf(rp[(h << 16) + d * EE + e]);
}

// ---------------------------------------------------------------------------
// Prep 2: codebooks [H][C][E] f32 -> l2-normalized bf16 [H][C][E]
// one wave per codebook row (E=64 -> 2 elems/lane)
// ---------------------------------------------------------------------------
__global__ void __launch_bounds__(256) prep_cb_kernel(const float* __restrict__ cb,
                                                      unsigned short* __restrict__ cbn) {
  const int lane = threadIdx.x & 31;
  const int row  = (blockIdx.x * 256 + threadIdx.x) >> 5;   // 0..H*C-1
  const float* r = cb + (size_t)row * EE;
  const float a = r[lane], b = r[lane + 32];
  float ss = a * a + b * b;
#pragma unroll
  for (int m = 1; m < 32; m <<= 1) ss += __shfl_xor(ss, m, 32);
  const float scale = 1.0f / fmaxf(sqrtf(ss), 1e-12f);
  cbn[(size_t)row * EE + lane]      = f2bf(a * scale);
  cbn[(size_t)row * EE + lane + 32] = f2bf(b * scale);
}

// ---------------------------------------------------------------------------
// Main fused kernel: LN -> projT WMMA -> simT WMMA -> argmax
// grid = 256 blocks x 256 threads; wave w handles head w for 64 tokens
// ---------------------------------------------------------------------------
__global__ void __launch_bounds__(256, 1)
rpq_main_kernel(const float* __restrict__ x,
                const unsigned short* __restrict__ wt_,
                const unsigned short* __restrict__ cbn_,
                float* __restrict__ out) {
  extern __shared__ char smem[];
  __bf16* s_xn = (__bf16*)smem;                         // [TOKW][XNS]

  const int tid  = threadIdx.x;
  const int lane = tid & 31;
  const int wv   = tid >> 5;            // wave id == head id
  const int cl   = lane & 15;           // column within 16-wide tile
  const bool hi  = lane >= 16;
  const int hiA  = hi ? 8 : 0;          // chunked A-frag K sub-offset (halves)
  const int hiB  = hi ? 16 : 0;         // contiguous frag K sub-offset (halves)
  const int rb   = hi ? 8 : 0;          // C/D row base for this lane half
  const long tok0 = (long)blockIdx.x * TOKW;

  // ---- Stage A: LayerNorm 8 tokens per wave, store bf16 xn tile to LDS ----
#pragma unroll
  for (int t = 0; t < TOKW / 8; ++t) {
    const int row = wv * (TOKW / 8) + t;                 // 0..TOKW-1
    const float* xr = x + (tok0 + row) * DD;
    float4 v[8];
    float s1 = 0.f, s2 = 0.f;
#pragma unroll
    for (int j = 0; j < 8; ++j) {
      v[j] = *(const float4*)(xr + j * 128 + lane * 4);
      s1 += v[j].x + v[j].y + v[j].z + v[j].w;
      s2 += v[j].x * v[j].x + v[j].y * v[j].y + v[j].z * v[j].z + v[j].w * v[j].w;
    }
#pragma unroll
    for (int m = 1; m < 32; m <<= 1) {
      s1 += __shfl_xor(s1, m, 32);
      s2 += __shfl_xor(s2, m, 32);
    }
    const float mean = s1 * (1.0f / DD);
    const float var  = s2 * (1.0f / DD) - mean * mean;
    const float rstd = rsqrtf(var + 1e-5f);
    __bf16* dst = s_xn + row * XNS;
#pragma unroll
    for (int j = 0; j < 8; ++j) {
      uint2 p;
      p.x = pk2bf((v[j].x - mean) * rstd, (v[j].y - mean) * rstd);
      p.y = pk2bf((v[j].z - mean) * rstd, (v[j].w - mean) * rstd);
      *(uint2*)(dst + j * 128 + lane * 4) = p;
    }
  }
  __syncthreads();

  // ---- Stage B: projT = W_h (A: ExD) x xn^T (B: DxM), head h = wv --------
  // D-layout: M-dim = e (4 tiles), N-dim = tokens (MSUB tiles)
  const int h = wv;
  const __bf16* wh = (const __bf16*)wt_ + (size_t)h * EE * DD;   // [E][D] bf16
  const v8f zero8 = {0.f, 0.f, 0.f, 0.f, 0.f, 0.f, 0.f, 0.f};
  v8f acc[4][MSUB];
#pragma unroll
  for (int t = 0; t < 4; ++t)
#pragma unroll
    for (int s = 0; s < MSUB; ++s) acc[t][s] = zero8;

  for (int k0 = 0; k0 < DD; k0 += 32) {
    v16bf afr[4];
#pragma unroll
    for (int t = 0; t < 4; ++t)
      afr[t] = ld_frag_a(wh + (t * 16 + cl) * DD + k0 + hiA);
    v16bf bfr[MSUB];
#pragma unroll
    for (int s = 0; s < MSUB; ++s)
      bfr[s] = ld_frag_a(s_xn + (s * 16 + cl) * XNS + k0 + hiA);
#pragma unroll
    for (int t = 0; t < 4; ++t)
#pragma unroll
      for (int s = 0; s < MSUB; ++s)
        acc[t][s] = wmma_bf16(afr[t], bfr[s], acc[t][s]);
  }

  // q-normalization is argmax-invariant (positive per-row scale) -> skipped.
  //
  // GEMM2 in standard K order (k = e - 32f, no permutation): for frag f, the K rows are
  //   K 0..7   -> e = 32f +  rb_lo(0)+0..7      (tile 2f,   own rows, lo lanes)
  //   K 8..15  -> e = 32f +  8..15              (tile 2f,   own rows, hi lanes)
  //   K 16..23 -> e = 32f + 16..23              (tile 2f+1, own rows, lo lanes)
  //   K 24..31 -> e = 32f + 24..31              (tile 2f+1, own rows, hi lanes)
  // => B-fragment is a pure per-lane repack of the accumulators (no shuffles),
  //    and the matching A-fragment read of cbn is two 16-byte chunks (ld_frag_a).
  v16bf bf2[MSUB][2];
#pragma unroll
  for (int s = 0; s < MSUB; ++s)
#pragma unroll
    for (int f = 0; f < 2; ++f) {
      union { unsigned u[8]; v16bf v; } u;
#pragma unroll
      for (int j = 0; j < 4; ++j) {
        u.u[j]     = pk2bf(acc[2 * f][s][2 * j],     acc[2 * f][s][2 * j + 1]);
        u.u[4 + j] = pk2bf(acc[2 * f + 1][s][2 * j], acc[2 * f + 1][s][2 * j + 1]);
      }
      bf2[s][f] = u.v;
    }

  // ---- Stage C: simT = cbn_h (A: CxE) x projT (B: ExM), fused argmax -----
  const __bf16* cbh = (const __bf16*)cbn_ + (size_t)h * CC * EE;  // [C][E] bf16
  float best[MSUB];
  int   bidx[MSUB];
#pragma unroll
  for (int s = 0; s < MSUB; ++s) { best[s] = -3.0e38f; bidx[s] = 0; }

  for (int c0 = 0; c0 < CC; c0 += 16) {
    v16bf a2[2];
#pragma unroll
    for (int f = 0; f < 2; ++f)   // standard-K A-frag: two 16B chunks, matching accumulator repack
      a2[f] = ld_frag_a(cbh + (c0 + cl) * EE + f * 32 + hiA);
#pragma unroll
    for (int s = 0; s < MSUB; ++s) {
      v8f sim = zero8;
      sim = wmma_bf16(a2[0], bf2[s][0], sim);
      sim = wmma_bf16(a2[1], bf2[s][1], sim);
      // lane holds token m = s*16+cl, codebook rows c0+rb+vv (ascending c,
      // so strict '>' keeps the first max, matching jnp.argmax).
#pragma unroll
      for (int vv = 0; vv < 8; ++vv) {
        if (sim[vv] > best[s]) { best[s] = sim[vv]; bidx[s] = c0 + rb + vv; }
      }
    }
  }

  // Merge the two lane halves (c = 16k+{0..7} vs 16k+{8..15}); tie -> min c.
#pragma unroll
  for (int s = 0; s < MSUB; ++s) {
    float bv = best[s];
    int   bi = bidx[s];
    const float ov = __shfl_xor(bv, 16, 32);
    const int   oi = __shfl_xor(bi, 16, 32);
    if (ov > bv || (ov == bv && oi < bi)) { bv = ov; bi = oi; }
    if (!hi) {
      const long tok = tok0 + s * 16 + cl;
      out[tok * HH + h] = (float)bi;
    }
  }
}

// ---------------------------------------------------------------------------
extern "C" void kernel_launch(void* const* d_in, const int* in_sizes, int n_in,
                              void* d_out, int out_size, void* d_ws, size_t ws_size,
                              hipStream_t stream) {
  (void)in_sizes; (void)n_in; (void)out_size; (void)ws_size;
  const float* x  = (const float*)d_in[0];
  const float* rp = (const float*)d_in[1];   // [H][D][E]
  const float* cb = (const float*)d_in[2];   // [H][C][E]

  unsigned short* wt  = (unsigned short*)d_ws;                         // 1 MiB bf16 [H][E][D]
  unsigned short* cbn = (unsigned short*)((char*)d_ws + (size_t)HH * EE * DD * 2);
  float* out = (float*)d_out;

  hipFuncSetAttribute(reinterpret_cast<const void*>(rpq_main_kernel),
                      hipFuncAttributeMaxDynamicSharedMemorySize, SMEM_MAIN);

  prep_w_kernel<<<(HH * EE * DD) / 256, 256, 0, stream>>>(rp, wt);
  prep_cb_kernel<<<(HH * CC) / 8, 256, 0, stream>>>(cb, cbn);

  const int n_tokens = 8 * 2048;
  rpq_main_kernel<<<n_tokens / TOKW, 256, SMEM_MAIN, stream>>>(x, wt, cbn, out);
}
